// MyRNN_51522427683061
// MI455X (gfx1250) — matrix operand order repaired and checked by reading, not compiled, not run on hardware
//
#include <hip/hip_runtime.h>
#include <cstdint>
#include <cstddef>

// ---------------------------------------------------------------------------
// 2-layer SimpleRNN for MI455X (gfx1250), wave32 + v_wmma_f32_16x16x32_f16.
//
//  Phase 0a: emb f32 -> f16 copy (gather source; 12.8MB, lives in 192MB L2)
//  Phase 0b: W1/U1/W2/U2 f32 -> f16, pre-swizzled into WMMA B-fragment order
//  Phase  A: Z1[t,b,:] = emb[inputs[b,t]] @ W1 + b1  (batched GEMM, double-
//            buffered gather staging, fp16 C-fragment output)
//  Phase  B: persistent recurrence kernel, 64 WGs x 8 waves, ping-pong h1/h2
//            in LDS, split WMMA accumulator chains, pipelined Z1 loads,
//            dscnt-only workgroup barriers, v_tanh_f32.
// ---------------------------------------------------------------------------

typedef __attribute__((ext_vector_type(16))) _Float16 v16h;
typedef __attribute__((ext_vector_type(8)))  _Float16 v8h;
typedef __attribute__((ext_vector_type(8)))  float    v8f;

#define VOCAB 50000
#define EMB   128
#define SEQ   256
#define UNITS 128
#define BATCH 1024

// workspace layout (bytes); total ~80.0 MB
#define EMBH_OFF   0
#define EMBH_BYTES ((size_t)VOCAB * EMB * 2)               // 12,800,000
#define FRAG_BYTES ((size_t)4 * 8 * 32 * 16 * 2)           // 32,768 per matrix
#define W1F_OFF    (EMBH_OFF + EMBH_BYTES)
#define U1F_OFF    (W1F_OFF + FRAG_BYTES)
#define W2F_OFF    (U1F_OFF + FRAG_BYTES)
#define U2F_OFF    (W2F_OFF + FRAG_BYTES)
#define Z1_OFF     (U2F_OFF + FRAG_BYTES)

// ---------------------------------------------------------------------------
// Light workgroup barrier: only orders LDS (dscnt), does NOT fence pending
// global loads/stores -> pipelined Z1 loads survive the barrier.
// ---------------------------------------------------------------------------
__device__ __forceinline__ void wg_barrier_lds() {
#if __has_builtin(__builtin_amdgcn_s_wait_dscnt) && \
    __has_builtin(__builtin_amdgcn_s_barrier_signal) && \
    __has_builtin(__builtin_amdgcn_s_barrier_wait)
  asm volatile("" ::: "memory");
  __builtin_amdgcn_s_wait_dscnt(0);
  __builtin_amdgcn_s_barrier_signal(-1);
  __builtin_amdgcn_s_barrier_wait((short)-1);
  asm volatile("" ::: "memory");
#else
  __syncthreads();
#endif
}

__device__ __forceinline__ float fast_tanh(float x) {
#if __has_builtin(__builtin_amdgcn_tanhf)
  return __builtin_amdgcn_tanhf(x);           // v_tanh_f32 (TRANS unit)
#elif __has_builtin(__builtin_amdgcn_tanh_f32)
  return __builtin_amdgcn_tanh_f32(x);
#else
  return tanhf(x);
#endif
}

// ---------------------------------------------------------------------------
// A-fragment loader: 16x32 f16 tile from a row-major [16][128] f16 LDS array.
// Per ISA 7.12.2: lane m=lane&15, hi=lane>>4;
//   frag[i]   (i<8)  <- k = kc*32 +      hi*8 + i
//   frag[i+8]        <- k = kc*32 + 16 + hi*8 + i
// => two contiguous 16B LDS runs per lane.
// ---------------------------------------------------------------------------
__device__ __forceinline__ v16h lds_a_frag(const _Float16* h, int kc, int lane) {
  const int m = lane & 15, hi = lane >> 4;
  const _Float16* p = h + m * 128 + kc * 32 + hi * 8;
  v8h lo = *(const v8h*)p;
  v8h hv = *(const v8h*)(p + 16);
  v16h r;
#pragma unroll
  for (int i = 0; i < 8; ++i) { r[i] = lo[i]; r[i + 8] = hv[i]; }
  return r;
}

#define WMMA(a, b, c) \
  __builtin_amdgcn_wmma_f32_16x16x32_f16(false, (a), false, (b), (short)0, (c), false, false)

// ---------------------------------------------------------------------------
// Phase 0a: embedding table f32 -> f16
// ---------------------------------------------------------------------------
__global__ __launch_bounds__(256) void k_prep_emb(const float* __restrict__ src,
                                                  _Float16* __restrict__ dst,
                                                  int n) {
  for (int i = blockIdx.x * 256 + threadIdx.x; i < n; i += gridDim.x * 256)
    dst[i] = (_Float16)src[i];
}

// ---------------------------------------------------------------------------
// Phase 0b: weights -> B-fragment order f16.
// B-fragment (16x16x32 f16): element i of lane L maps to
//   k = kc*32 + 16*(i>=8) + 8*(L>=16) + (i&7),   n = nt*16 + (L&15)
// Storage: [matrix][kc][nt][lane][i] contiguous (32B per lane per fragment).
// ---------------------------------------------------------------------------
__global__ __launch_bounds__(256) void k_prep_wfrag(const float* __restrict__ W1,
                                                    const float* __restrict__ U1,
                                                    const float* __restrict__ W2,
                                                    const float* __restrict__ U2,
                                                    _Float16* __restrict__ dst) {
  const float* srcs[4] = {W1, U1, W2, U2};
  const int total = 4 * 16384;
  for (int e = blockIdx.x * 256 + threadIdx.x; e < total; e += gridDim.x * 256) {
    int mtx  = e >> 14;
    int r    = e & 16383;             // r = kc*4096 + nt*512 + lane*16 + i
    int i    = r & 15;
    int lane = (r >> 4) & 31;
    int nt   = (r >> 9) & 7;
    int kc   = (r >> 12) & 3;
    int k = kc * 32 + ((i >> 3) << 4) + ((lane >> 4) << 3) + (i & 7);
    int n = nt * 16 + (lane & 15);
    dst[(size_t)mtx * 16384 + r] = (_Float16)srcs[mtx][k * 128 + n];
  }
}

// ---------------------------------------------------------------------------
// Phase A: Z1 = gather(emb_f16, inputs) @ W1 + b1, fp16 C-fragment output.
// 2048 blocks x 256 threads; each block does 8 M-strips with a double-
// buffered gather stage (strip i+1's gather overlaps strip i's WMMAs).
// M index: mt = t*64 + batch_strip, mt in [0, 16384).
// ---------------------------------------------------------------------------
__global__ __launch_bounds__(256) void k_phaseA(const int* __restrict__ inp,
                                                const _Float16* __restrict__ embh,
                                                const _Float16* __restrict__ w1f,
                                                const float* __restrict__ b1,
                                                _Float16* __restrict__ z1) {
  __shared__ _Float16 As[2][16 * 128];
  const int tid = threadIdx.x;
  const int w = tid >> 5, lane = tid & 31, m = lane & 15;
  const int rr = tid >> 4, cc = tid & 15;
  const float b1v = b1[w * 16 + m];

  v16h bf[4];
#pragma unroll
  for (int kc = 0; kc < 4; ++kc)
    bf[kc] = *(const v16h*)(w1f + (((kc << 3) + w) * 32 + lane) * 16);

  // preload all 8 gather indices (independent b32 loads)
  int idxv[8];
#pragma unroll
  for (int it = 0; it < 8; ++it) {
    const int mt = blockIdx.x * 8 + it;
    idxv[it] = inp[((((mt & 63) << 4) + rr) * SEQ) + (mt >> 6)];
  }

  // stage strip 0
  *(v8h*)(&As[0][rr * 128 + cc * 8]) =
      *(const v8h*)(embh + (size_t)idxv[0] * EMB + cc * 8);
  wg_barrier_lds();

  int cur = 0;
#pragma unroll
  for (int it = 0; it < 8; ++it) {
    v8h stage;
    if (it < 7)  // issue next strip's gather before compute (latency hiding)
      stage = *(const v8h*)(embh + (size_t)idxv[it + 1] * EMB + cc * 8);

    const int mt = blockIdx.x * 8 + it;
    v8f accA, accB;
#pragma unroll
    for (int i = 0; i < 8; ++i) { accA[i] = b1v; accB[i] = 0.f; }
    accA = WMMA(lds_a_frag(As[cur], 0, lane), bf[0], accA);
    accB = WMMA(lds_a_frag(As[cur], 1, lane), bf[1], accB);
    accA = WMMA(lds_a_frag(As[cur], 2, lane), bf[2], accA);
    accB = WMMA(lds_a_frag(As[cur], 3, lane), bf[3], accB);

    v8h zc;
#pragma unroll
    for (int i = 0; i < 8; ++i) zc[i] = (_Float16)(accA[i] + accB[i]);
    *(v8h*)(z1 + (((size_t)mt * 8 + w) * 32 + lane) * 8) = zc;

    if (it < 7)
      *(v8h*)(&As[cur ^ 1][rr * 128 + cc * 8]) = stage;
    wg_barrier_lds();
    cur ^= 1;
  }
}

// ---------------------------------------------------------------------------
// Phase B: fused 2-layer recurrence + final FC. 64 WGs, one per 16-row batch
// strip; wave w owns output columns [16w, 16w+16). Weights in 96 VGPRs/lane,
// ping-pong h1/h2 in LDS (2 barriers/step), pipelined Z1 loads, split WMMA
// accumulator chains for ILP.
// ---------------------------------------------------------------------------
__global__ __launch_bounds__(256) void k_phaseB(const _Float16* __restrict__ z1,
                                                const _Float16* __restrict__ u1f,
                                                const _Float16* __restrict__ w2f,
                                                const _Float16* __restrict__ u2f,
                                                const float* __restrict__ b2,
                                                const float* __restrict__ fcw,
                                                const float* __restrict__ fcb,
                                                float* __restrict__ out) {
  __shared__ _Float16 h1s[2][16 * 128];
  __shared__ _Float16 h2s[2][16 * 128];
  const int tid = threadIdx.x;
  const int w = tid >> 5, lane = tid & 31, m = lane & 15, hi = lane >> 4;
  const int s = blockIdx.x;                       // batch strip [s*16, s*16+16)
  const float b2v = b2[w * 16 + m];

  v16h u1[4], w2[4], u2[4];
#pragma unroll
  for (int kc = 0; kc < 4; ++kc) {
    const size_t fo = (size_t)(((kc << 3) + w) * 32 + lane) * 16;
    u1[kc] = *(const v16h*)(u1f + fo);
    w2[kc] = *(const v16h*)(w2f + fo);
    u2[kc] = *(const v16h*)(u2f + fo);
  }
  for (int i = tid; i < 16 * 128; i += 256) {
    h1s[0][i] = (_Float16)0.f; h1s[1][i] = (_Float16)0.f;
    h2s[0][i] = (_Float16)0.f; h2s[1][i] = (_Float16)0.f;
  }
  __syncthreads();

  // pipelined Z1 fragment: load t=0 before the loop
  v8h zc = *(const v8h*)(z1 + (((size_t)s * 8 + w) * 32 + lane) * 8);

  int cur = 0;
  for (int t = 0; t < SEQ; ++t) {
    const int nxt = cur ^ 1;

    // ---- layer 1: acc1 = Z1 + h1_old @ U1 (two independent WMMA chains)
    v8f a1a, a1b;
#pragma unroll
    for (int i = 0; i < 8; ++i) { a1a[i] = (float)zc[i]; a1b[i] = 0.f; }
    a1a = WMMA(lds_a_frag(h1s[cur], 0, lane), u1[0], a1a);
    a1b = WMMA(lds_a_frag(h1s[cur], 1, lane), u1[1], a1b);
    a1a = WMMA(lds_a_frag(h1s[cur], 2, lane), u1[2], a1a);
    a1b = WMMA(lds_a_frag(h1s[cur], 3, lane), u1[3], a1b);

    // issue next step's Z1 load now; dscnt-only barriers won't stall on it
    if (t + 1 < SEQ)
      zc = *(const v8h*)(z1 + ((((size_t)(t + 1) * 64 + s) * 8 + w) * 32 + lane) * 8);

    float ht[8];
#pragma unroll
    for (int i = 0; i < 8; ++i) ht[i] = fast_tanh(a1a[i] + a1b[i]);
#pragma unroll
    for (int i = 0; i < 8; ++i)                       // C-frag: row = i + hi*8
      h1s[nxt][(i + hi * 8) * 128 + w * 16 + m] = (_Float16)ht[i];
    wg_barrier_lds();                                 // h1_new visible

    // ---- layer 2: acc2 = b2 + h1_new @ W2 + h2_old @ U2 (4 chains, depth 2)
    v8f aW0, aW1, aU0, aU1;
#pragma unroll
    for (int i = 0; i < 8; ++i) { aW0[i] = b2v; aW1[i] = 0.f; aU0[i] = 0.f; aU1[i] = 0.f; }
    aW0 = WMMA(lds_a_frag(h1s[nxt], 0, lane), w2[0], aW0);
    aW1 = WMMA(lds_a_frag(h1s[nxt], 1, lane), w2[1], aW1);
    aU0 = WMMA(lds_a_frag(h2s[cur], 0, lane), u2[0], aU0);
    aU1 = WMMA(lds_a_frag(h2s[cur], 1, lane), u2[1], aU1);
    aW0 = WMMA(lds_a_frag(h1s[nxt], 2, lane), w2[2], aW0);
    aW1 = WMMA(lds_a_frag(h1s[nxt], 3, lane), w2[3], aW1);
    aU0 = WMMA(lds_a_frag(h2s[cur], 2, lane), u2[2], aU0);
    aU1 = WMMA(lds_a_frag(h2s[cur], 3, lane), u2[3], aU1);
#pragma unroll
    for (int i = 0; i < 8; ++i)
      ht[i] = fast_tanh((aW0[i] + aW1[i]) + (aU0[i] + aU1[i]));
#pragma unroll
    for (int i = 0; i < 8; ++i)
      h2s[nxt][(i + hi * 8) * 128 + w * 16 + m] = (_Float16)ht[i];
    wg_barrier_lds();                                 // h2_new visible

    cur = nxt;
  }

  // final: out[b] = sigmoid(h2_last @ fc_w + fc_b); wave 0, lanes 0-15
  if (w == 0 && lane < 16) {
    float sum = fcb[0];
    for (int c = 0; c < UNITS; ++c)
      sum += (float)h2s[cur][lane * 128 + c] * fcw[c];
    out[s * 16 + lane] = 1.0f / (1.0f + expf(-sum));
  }
}

// ---------------------------------------------------------------------------
extern "C" void kernel_launch(void* const* d_in, const int* in_sizes, int n_in,
                              void* d_out, int out_size, void* d_ws, size_t ws_size,
                              hipStream_t stream) {
  (void)in_sizes; (void)n_in; (void)out_size; (void)ws_size;

  const int*   inp  = (const int*)d_in[0];
  const float* emb  = (const float*)d_in[1];
  const float* W1   = (const float*)d_in[2];
  const float* U1   = (const float*)d_in[3];
  const float* b1   = (const float*)d_in[4];
  const float* W2   = (const float*)d_in[5];
  const float* U2   = (const float*)d_in[6];
  const float* b2   = (const float*)d_in[7];
  const float* fcw  = (const float*)d_in[8];
  const float* fcb  = (const float*)d_in[9];
  float*       out  = (float*)d_out;

  char* ws = (char*)d_ws;
  _Float16* embh = (_Float16*)(ws + EMBH_OFF);
  _Float16* w1f  = (_Float16*)(ws + W1F_OFF);
  _Float16* u1f  = (_Float16*)(ws + U1F_OFF);
  _Float16* w2f  = (_Float16*)(ws + W2F_OFF);
  _Float16* u2f  = (_Float16*)(ws + U2F_OFF);
  _Float16* z1   = (_Float16*)(ws + Z1_OFF);

  k_prep_emb<<<2048, 256, 0, stream>>>(emb, embh, VOCAB * EMB);
  k_prep_wfrag<<<256, 256, 0, stream>>>(W1, U1, W2, U2, w1f);
  k_phaseA<<<2048, 256, 0, stream>>>(inp, embh, w1f, b1, z1);
  k_phaseB<<<64, 256, 0, stream>>>(z1, u1f, w2f, u2f, b2, fcw, fcb, out);
}
